// ModuleCorrelation_5205500363444
// MI455X (gfx1250) — matrix operand order repaired and checked by reading, not compiled, not run on hardware
//
#include <hip/hip_runtime.h>

typedef _Float16 v16h __attribute__((ext_vector_type(16)));
typedef _Float16 v8h  __attribute__((ext_vector_type(8)));
typedef float    v8f  __attribute__((ext_vector_type(8)));

namespace {
constexpr int kB = 8, kC = 128, kH = 192, kW = 448;
constexpr int kKC   = 32;              // channels per WMMA K-chunk
constexpr int kNKC  = kC / kKC;        // 4 chunks
constexpr int kStrip = 64;             // output w-columns per workgroup (4 waves x 16)
constexpr int kWin   = 80;             // staged 'second' window: strip + 16
constexpr int kRow   = 40;             // padded LDS row in halves (32 data + 8 pad = 80B)
constexpr int kDJ    = 7;
constexpr int kThreads = 128;          // 4 wave32s

constexpr int kOffAhi = 0;
constexpr int kOffAlo = kOffAhi + kStrip * kRow * 2;           // 5120
constexpr int kOffBhi = kOffAlo + kStrip * kRow * 2;           // 10240
constexpr int kOffBlo = kOffBhi + kDJ * kWin * kRow * 2;       // 55040
constexpr int kLdsBytes = kOffBlo + kDJ * kWin * kRow * 2;     // 99840 (< 320KB WGP LDS)
}

union Frag16 { v16h v; v8h h[2]; };

// Correlation (49 displacements, stride 1) via banded 16x16x32 f16 WMMA with
// hi/lo fp16 splitting (3 WMMAs per tile) for fp32-grade accuracy.
__global__ __launch_bounds__(kThreads) void corr49_wmma_f16x3(
    const float* __restrict__ first, const float* __restrict__ second,
    float* __restrict__ out)
{
  __shared__ __align__(16) unsigned char smem[kLdsBytes];
  _Float16* sAhi = reinterpret_cast<_Float16*>(smem + kOffAhi);
  _Float16* sAlo = reinterpret_cast<_Float16*>(smem + kOffAlo);
  _Float16* sBhi = reinterpret_cast<_Float16*>(smem + kOffBhi);
  _Float16* sBlo = reinterpret_cast<_Float16*>(smem + kOffBlo);

  const int tid  = threadIdx.x;
  const int lane = tid & 31;
  const int ln   = lane & 15;
  const int hiH  = lane >> 4;            // 0: lanes 0-15, 1: lanes 16-31
  const int aw   = (tid >> 5) << 4;      // this wave's 16-col tile base in strip

  int blk = blockIdx.x;
  const int ws = blk % 7; blk /= 7;      // 448 / 64 = 7 strips
  const int h  = blk % kH;
  const int bb = blk / kH;
  const int w0 = ws * kStrip;

  const size_t planeHW = (size_t)kH * kW;
  const float* F = first  + (size_t)bb * kC * planeHW;
  const float* S = second + (size_t)bb * kC * planeHW;

  v8f acc[kDJ][2];
  const v8f vzero = {0.f, 0.f, 0.f, 0.f, 0.f, 0.f, 0.f, 0.f};
#pragma unroll
  for (int d = 0; d < kDJ; ++d) { acc[d][0] = vzero; acc[d][1] = vzero; }

  for (int kc = 0; kc < kNKC; ++kc) {
    const size_t cbase = (size_t)kc * kKC * planeHW;

    // ---- stage A: first[b, kc*32..+31, h, w0..w0+63] -> f16 hi/lo, [w][c] layout
    for (int e = tid; e < kKC * kStrip; e += kThreads) {
      const int c = e >> 6, w = e & 63;                 // consecutive tid -> consecutive w
      const float x = F[cbase + (size_t)c * planeHW + (size_t)h * kW + (w0 + w)];
      const _Float16 hi = (_Float16)x;
      const _Float16 lo = (_Float16)(x - (float)hi);
      sAhi[w * kRow + c] = hi;
      sAlo[w * kRow + c] = lo;
    }
    // ---- stage B: second[b, c, h+dj-3, w0-3 .. w0+76], zeros outside image
    for (int e = tid; e < kDJ * kKC * kWin; e += kThreads) {
      const int r  = e / kWin;                          // (dj, c)
      const int w  = e - r * kWin;                      // window column 0..79
      const int dj = r >> 5, c = r & 31;
      const int hr = h + dj - 3;
      const int wc = w0 - 3 + w;
      float x = 0.0f;
      if ((unsigned)hr < (unsigned)kH && (unsigned)wc < (unsigned)kW)
        x = S[cbase + (size_t)c * planeHW + (size_t)hr * kW + wc];
      const _Float16 hi = (_Float16)x;
      const _Float16 lo = (_Float16)(x - (float)hi);
      sBhi[(dj * kWin + w) * kRow + c] = hi;
      sBlo[(dj * kWin + w) * kRow + c] = lo;
    }
    __syncthreads();

    // ---- A fragments (ISA 16-bit A 16x32 layout: lanes<16 K{0..7,16..23},
    //      lanes>=16 K{8..15,24..31}); two ds_load_b128 per half
    Frag16 ahi, alo;
    {
      const int ar = (aw + ln) * kRow + hiH * 8;
      ahi.h[0] = *(const v8h*)(sAhi + ar);
      ahi.h[1] = *(const v8h*)(sAhi + ar + 16);
      alo.h[0] = *(const v8h*)(sAlo + ar);
      alo.h[1] = *(const v8h*)(sAlo + ar + 16);
    }
#pragma unroll
    for (int dj = 0; dj < kDJ; ++dj) {
#pragma unroll
      for (int bt = 0; bt < 2; ++bt) {
        // B 32x16 layout: lane n<16 = col n, K{0..7,16..23}; lane n+16 = col n, K{8..15,24..31}
        Frag16 bhi, blo;
        const int br = (dj * kWin + aw + bt * 16 + ln) * kRow + hiH * 8;
        bhi.h[0] = *(const v8h*)(sBhi + br);
        bhi.h[1] = *(const v8h*)(sBhi + br + 16);
        blo.h[0] = *(const v8h*)(sBlo + br);
        blo.h[1] = *(const v8h*)(sBlo + br + 16);
        // fp32-grade product via 3-term f16 split: hi*hi + hi*lo + lo*hi
        acc[dj][bt] = __builtin_amdgcn_wmma_f32_16x16x32_f16(
            false, ahi.v, false, bhi.v, (short)0, acc[dj][bt], false, false);
        acc[dj][bt] = __builtin_amdgcn_wmma_f32_16x16x32_f16(
            false, ahi.v, false, blo.v, (short)0, acc[dj][bt], false, false);
        acc[dj][bt] = __builtin_amdgcn_wmma_f32_16x16x32_f16(
            false, alo.v, false, bhi.v, (short)0, acc[dj][bt], false, false);
      }
    }
    __syncthreads();
  }

  // ---- band diagonals -> (49 x 64) tile in LDS, then coalesced global store.
  // C/D layout: VGPR v, lanes<16 -> (M=v, N=lane); lanes>=16 -> (M=v+8, N=lane-16)
  float* Dout = reinterpret_cast<float*>(smem);   // reuse staging LDS (12544 B)
#pragma unroll
  for (int dj = 0; dj < kDJ; ++dj) {
#pragma unroll
    for (int bt = 0; bt < 2; ++bt) {
#pragma unroll
      for (int v = 0; v < 8; ++v) {
        const int m  = v + hiH * 8;
        const int ng = ln + bt * 16;
        const int di = ng - m;                    // displacement index 0..6 when valid
        if (di >= 0 && di < 7)
          Dout[(dj * 7 + di) * kStrip + aw + m] = acc[dj][bt][v] * (1.0f / 128.0f);
      }
    }
  }
  __syncthreads();
  for (int e = tid; e < 49 * kStrip; e += kThreads) {
    const int d = e >> 6, w = e & 63;
    out[(((size_t)bb * 49 + d) * kH + h) * kW + (w0 + w)] = Dout[e];
  }
}

extern "C" void kernel_launch(void* const* d_in, const int* in_sizes, int n_in,
                              void* d_out, int out_size, void* d_ws, size_t ws_size,
                              hipStream_t stream) {
  (void)in_sizes; (void)n_in; (void)d_ws; (void)ws_size; (void)out_size;
  const float* first  = (const float*)d_in[0];
  const float* second = (const float*)d_in[1];
  // d_in[2] = intStride == 1 for this problem; kernel specializes stride 1.
  float* out = (float*)d_out;
  dim3 grid(kB * kH * (kW / kStrip));   // 8 * 192 * 7 = 10752 workgroups
  dim3 block(kThreads);                 // 4 wave32s
  hipLaunchKernelGGL(corr49_wmma_f16x3, grid, block, 0, stream, first, second, out);
}
